// CommandScorerWithDN_5772436046386
// MI455X (gfx1250) — matrix operand order repaired and checked by reading, not compiled, not run on hardware
//
#include <hip/hip_runtime.h>
#include <math.h>

typedef _Float16 h16;
typedef __attribute__((ext_vector_type(16))) _Float16 v16h;
typedef __attribute__((ext_vector_type(8)))  _Float16 v8h;
typedef __attribute__((ext_vector_type(8)))  float    v8f;

#define DIV_UP(a,b) (((a)+(b)-1)/(b))

__device__ __forceinline__ float sigmoidf_(float x){ return 1.f/(1.f+__expf(-x)); }
__device__ __forceinline__ float eluf_(float x){ return x > 0.f ? x : (__expf(x)-1.f); }

// Async global->LDS copy (CDNA5). ldsOff is the 32-bit LDS byte offset
// (low 32 bits of the flat shared-pointer, per the LDS aperture rule).
__device__ __forceinline__ void async_b128(unsigned ldsOff, const void* g){
  asm volatile("global_load_async_to_lds_b128 %0, %1, off"
               :: "v"(ldsOff), "v"(g) : "memory");
}
__device__ __forceinline__ void wait_async0(){
  asm volatile("s_wait_asynccnt 0x0" ::: "memory");
}

// ================= WMMA GEMM: C[M,N] = act(A[M,K] @ B[N,K]^T + bias) =======
// A rows optionally gathered: row m comes from A + gidx[m*gstride]*lda.
#define BM 128
#define BN 128
#define BK 32
#define LDSB 40   // padded LDS row stride (h16): 80B

__launch_bounds__(256)
__global__ void k_gemm_f16(const h16* __restrict__ A, int lda,
                           const int* __restrict__ gidx, int gstride,
                           const h16* __restrict__ B, int ldb,
                           float* __restrict__ C, int ldc,
                           h16* __restrict__ C2, int ldc2,
                           int M, int N, int K,
                           const float* __restrict__ bias, int act)
{
  __shared__ h16 As[2][BM*LDSB];
  __shared__ h16 Bs[2][BN*LDSB];

  const int tid  = threadIdx.x;
  const int wave = tid >> 5, lane = tid & 31;
  const int lm = lane & 15, lh = lane >> 4;
  const int wm = (wave & 3) * 32;   // 4 waves along M: 2 frags each
  const int wn = (wave >> 2) * 64;  // 2 waves along N: 4 frags each
  const int tileM = blockIdx.y * BM, tileN = blockIdx.x * BN;

  v8f acc[2][4] = {};

  const int row = tid >> 1;         // 0..127 (loader row for A and B tiles)
  const int ch  = tid & 1;          // which 16-elem chunk of the 32-wide row
  const bool fullA = (tileM + BM <= M);
  const bool fullB = (tileN + BN <= N);

  const int KT = K / BK;

  auto issueTile = [&](int kt, int buf) {
    const int k0 = kt * BK;
    // ---- A tile: BM x 32 ----
    {
      unsigned lds = (unsigned)(uintptr_t)&As[buf][row*LDSB + ch*16];
      if (fullA) {
        long r = gidx ? (long)gidx[(long)(tileM + row) * gstride] : (long)(tileM + row);
        const h16* g = A + r*(long)lda + k0 + ch*16;
        async_b128(lds,      (const void*)g);
        async_b128(lds + 16, (const void*)(g + 8));
      } else {
        int m = tileM + row;
        uint4 v0 = {0,0,0,0}, v1 = {0,0,0,0};
        if (m < M) {
          long r = gidx ? (long)gidx[(long)m * gstride] : (long)m;
          const uint4* s = (const uint4*)(A + r*(long)lda + k0 + ch*16);
          v0 = s[0]; v1 = s[1];
        }
        uint4* d = (uint4*)(&As[buf][row*LDSB + ch*16]);
        d[0] = v0; d[1] = v1;
      }
    }
    // ---- B tile: BN x 32 ----
    {
      unsigned lds = (unsigned)(uintptr_t)&Bs[buf][row*LDSB + ch*16];
      if (fullB) {
        const h16* g = B + (long)(tileN + row)*ldb + k0 + ch*16;
        async_b128(lds,      (const void*)g);
        async_b128(lds + 16, (const void*)(g + 8));
      } else {
        int n = tileN + row;
        uint4 v0 = {0,0,0,0}, v1 = {0,0,0,0};
        if (n < N) {
          const uint4* s = (const uint4*)(B + (long)n*ldb + k0 + ch*16);
          v0 = s[0]; v1 = s[1];
        }
        uint4* d = (uint4*)(&Bs[buf][row*LDSB + ch*16]);
        d[0] = v0; d[1] = v1;
      }
    }
  };

  issueTile(0, 0);
  wait_async0();
  __syncthreads();

  for (int kt = 0; kt < KT; ++kt) {
    const int buf = kt & 1;
    if (kt + 1 < KT) issueTile(kt + 1, buf ^ 1);

    v16h afr[2], bfr[4];
#pragma unroll
    for (int i = 0; i < 2; ++i) {
      // A 16x32 frag (ISA layout): elems 0..7 <- K=lh*8.., elems 8..15 <- K=16+lh*8..
      const h16* ap = &As[buf][(wm + i*16 + lm) * LDSB + lh*8];
      v8h alo = *(const v8h*)ap;
      v8h ahi = *(const v8h*)(ap + 16);
#pragma unroll
      for (int e = 0; e < 8; ++e) { afr[i][e] = alo[e]; afr[i][8+e] = ahi[e]; }
    }
#pragma unroll
    for (int j = 0; j < 4; ++j) {
      // B 32x16 frag: lane holds col n=lm, elems 0..7 <- K=lh*8.., elems 8..15 <- K=16+lh*8..
      const h16* bp = &Bs[buf][(wn + j*16 + lm) * LDSB + lh*8];
      v8h blo = *(const v8h*)bp;
      v8h bhi = *(const v8h*)(bp + 16);
#pragma unroll
      for (int e = 0; e < 8; ++e) { bfr[j][e] = blo[e]; bfr[j][8+e] = bhi[e]; }
    }
#pragma unroll
    for (int i = 0; i < 2; ++i)
#pragma unroll
      for (int j = 0; j < 4; ++j)
        acc[i][j] = __builtin_amdgcn_wmma_f32_16x16x32_f16(
            false, afr[i], false, bfr[j], (short)0, acc[i][j], false, false);

    wait_async0();
    __syncthreads();
  }

#pragma unroll
  for (int i = 0; i < 2; ++i)
#pragma unroll
    for (int j = 0; j < 4; ++j)
#pragma unroll
      for (int r = 0; r < 8; ++r) {
        int m = tileM + wm + i*16 + lh*8 + r;
        int n = tileN + wn + j*16 + lm;
        if (m < M && n < N) {
          float v = acc[i][j][r];
          if (bias) v += bias[n];
          if (act == 1) v = eluf_(v);
          if (C)  C[(long)m*ldc + n] = v;
          if (C2) C2[(long)m*ldc2 + n] = (h16)v;
        }
      }
}

// ========================= elementwise / utility kernels ====================
__global__ void k_zero_f32(float* p, long n){
  long i = (long)blockIdx.x*blockDim.x + threadIdx.x; if (i < n) p[i] = 0.f; }
__global__ void k_zero_h16(h16* p, long n){
  long i = (long)blockIdx.x*blockDim.x + threadIdx.x; if (i < n) p[i] = (h16)0.f; }

// out[r*out_ld + c] = in[r*in_ld + in_off + c]   (f32 -> f16 slice convert)
__global__ void k_conv_stride(const float* __restrict__ in, long in_ld, long in_off,
                              h16* __restrict__ out, long out_ld, long rows, long cols){
  long i = (long)blockIdx.x*blockDim.x + threadIdx.x;
  if (i >= rows*cols) return;
  long r = i / cols, c = i % cols;
  out[r*out_ld + c] = (h16)in[r*in_ld + in_off + c];
}

// out[c*rows + r] = in[r*cols + c] + addc
__global__ void k_transpose(const float* __restrict__ in, h16* __restrict__ out,
                            int rows, int cols, float addc){
  long i = (long)blockIdx.x*blockDim.x + threadIdx.x;
  if (i >= (long)rows*cols) return;
  int r = i / cols, c = i % cols;
  out[(long)c*rows + r] = (h16)(in[i] + addc);
}

__global__ void k_emb_pad(const float* __restrict__ emb, h16* __restrict__ out){
  long i = (long)blockIdx.x*blockDim.x + threadIdx.x;
  if (i >= 20000L*320) return;
  long v = i / 320, c = i % 320;
  out[i] = (c < 300) ? (h16)emb[v*300 + c] : (h16)0.f;
}

__global__ void k_pack_ids(const int* __restrict__ e, const int* __restrict__ l,
                           const int* __restrict__ w, const int* __restrict__ c,
                           int* __restrict__ out){
  long i = (long)blockIdx.x*blockDim.x + threadIdx.x;
  if (i >= 180224) return;
  int v;
  if      (i < 16384)  v = e[i];
  else if (i < 32768)  v = l[i - 16384];
  else if (i < 163840) v = w[i - 32768];
  else                 v = c[i - 163840];
  out[i] = v;
}

// GRU cell: r=sig(xr+hr); z=sig(xz+hz); n=tanh(xn+r*hn); h'=(1-z)n+z h
__global__ void k_gru_update(const float* __restrict__ Gh, int ghld,
                             const float* __restrict__ Gx, int gxld,
                             int gxmul, int gxoff,
                             float* __restrict__ h, h16* __restrict__ hh,
                             int S, int HC){
  long idx = (long)blockIdx.x*blockDim.x + threadIdx.x;
  if (idx >= (long)S*HC) return;
  int s = idx / HC, j = idx % HC;
  const float* gx = Gx + ((long)s*gxmul + gxoff)*gxld;
  const float* gh = Gh + (long)s*ghld;
  float r = sigmoidf_(gx[j]      + gh[j]);
  float z = sigmoidf_(gx[HC+j]   + gh[HC+j]);
  float n = tanhf    (gx[2*HC+j] + r*gh[2*HC+j]);
  float nh = (1.f - z)*n + z*h[idx];
  h[idx] = nh;  hh[idx] = (h16)nh;
}

__global__ void k_concat_h(const float* __restrict__ hf, const float* __restrict__ hb,
                           h16* __restrict__ out){
  long i = (long)blockIdx.x*blockDim.x + threadIdx.x;
  if (i >= 11264L*512) return;
  long s = i / 512, j = i % 512;
  float v = (j < 256) ? hf[s*256 + j] : hb[s*256 + j - 256];
  out[i] = (h16)v;
}

__global__ void k_wsum(const h16* __restrict__ Hout, h16* __restrict__ out){
  long i = (long)blockIdx.x*blockDim.x + threadIdx.x;
  if (i >= 1024L*512) return;
  long r = i / 512, j = i % 512;
  float s = 0.f;
  for (int w = 0; w < 8; ++w) s += (float)Hout[(2048 + r*8 + w)*512 + j];
  out[i] = (h16)s;
}

__global__ void k_combine(const float* __restrict__ t1, const float* __restrict__ t2,
                          const float* __restrict__ t3, h16* __restrict__ out){
  long i = (long)blockIdx.x*blockDim.x + threadIdx.x;
  if (i >= 1024L*512) return;
  float gh = eluf_(t1[i]) + eluf_(t2[i]) + t3[i];
  out[i] = (h16)eluf_(gh);
}

// per (b,n): partial[b,n,c] = sum_j elu(g[b,n,j]+c[b,c,j]+ba1[j]) * Wa2[j]
__launch_bounds__(256)
__global__ void k_score(const float* __restrict__ cpart, const float* __restrict__ gpart,
                        const float* __restrict__ ba1, const float* __restrict__ Wa2,
                        float* __restrict__ partial){
  __shared__ float g[1024];
  __shared__ float red[256];
  int blk = blockIdx.x, tid = threadIdx.x;     // blk = b*64 + n
  const float* grow = gpart + (long)blk*1024;
  for (int j = tid; j < 1024; j += 256) g[j] = grow[j];
  __syncthreads();
  int b = blk >> 6;
  for (int c = 0; c < 64; ++c) {
    const float* crow = cpart + (long)(b*64 + c)*1024;
    float s = 0.f;
    for (int j = tid; j < 1024; j += 256)
      s += eluf_(g[j] + crow[j] + ba1[j]) * Wa2[j];
    red[tid] = s; __syncthreads();
    for (int o = 128; o > 0; o >>= 1) { if (tid < o) red[tid] += red[tid+o]; __syncthreads(); }
    if (tid == 0) partial[(long)blk*64 + c] = red[0];
    __syncthreads();
  }
}

__global__ void k_score_reduce(const float* __restrict__ partial,
                               const float* __restrict__ ba2, float* __restrict__ scores){
  int i = blockIdx.x*blockDim.x + threadIdx.x;
  if (i >= 1024) return;
  int b = i >> 6, c = i & 63;
  float s = 0.f;
  for (int n = 0; n < 64; ++n) s += partial[(long)(b*64+n)*64 + c];
  scores[i] = s * (1.f/64.f) + ba2[0];
}

__global__ void k_value(const float* __restrict__ hf, const float* __restrict__ hb,
                        const float* __restrict__ Wc, const float* __restrict__ bc,
                        float* __restrict__ out){
  __shared__ float red[256];
  int b = blockIdx.x, tid = threadIdx.x;
  float s = 0.f;
  for (int j = tid; j < 1024; j += 256) {
    float v = (j < 512) ? hf[b*512 + j] : hb[b*512 + j - 512];
    s += v * Wc[j];
  }
  red[tid] = s; __syncthreads();
  for (int o = 128; o > 0; o >>= 1) { if (tid < o) red[tid] += red[tid+o]; __syncthreads(); }
  if (tid == 0) out[b] = red[0] + bc[0];
}

__global__ void k_index(const float* __restrict__ scores, const int* __restrict__ commands,
                        float* __restrict__ outIdx){
  __shared__ float sv[64];
  int b = blockIdx.x, c = threadIdx.x;
  long sum = 0;
  for (int l = 0; l < 16; ++l) sum += commands[(b*64 + c)*16 + l];
  sv[c] = (sum > 0) ? scores[b*64 + c] : -1e9f;
  __syncthreads();
  if (c == 0) {
    int best = 0; float bv = sv[0];
    for (int i = 1; i < 64; ++i) if (sv[i] > bv) { bv = sv[i]; best = i; }
    outIdx[b] = (float)best;
  }
}

// =============================== host driver ===============================
extern "C" void kernel_launch(void* const* d_in, const int* in_sizes, int n_in,
                              void* d_out, int out_size, void* d_ws, size_t ws_size,
                              hipStream_t stream)
{
  (void)in_sizes; (void)n_in; (void)out_size; (void)ws_size;
  const float* emb   = (const float*)d_in[0];
  const float* Wprj  = (const float*)d_in[1];
  const float* dnWif = (const float*)d_in[2];
  const float* dnWhf = (const float*)d_in[3];
  const float* dnbif = (const float*)d_in[4];
  const float* dnbhf = (const float*)d_in[5];
  const float* dnWib = (const float*)d_in[6];
  const float* dnWhb = (const float*)d_in[7];
  const float* dnbib = (const float*)d_in[8];
  const float* dnbhb = (const float*)d_in[9];
  const float* W_e   = (const float*)d_in[10];
  const float* W_l   = (const float*)d_in[11];
  const float* W_w   = (const float*)d_in[12];
  const float* Wg    = (const float*)d_in[13];
  const float* bg    = (const float*)d_in[14];
  const float* dWif  = (const float*)d_in[15];
  const float* dWhf  = (const float*)d_in[16];
  const float* dbif  = (const float*)d_in[17];
  const float* dbhf  = (const float*)d_in[18];
  const float* dWib  = (const float*)d_in[19];
  const float* dWhb  = (const float*)d_in[20];
  const float* dbib  = (const float*)d_in[21];
  const float* dbhb  = (const float*)d_in[22];
  const float* Wc    = (const float*)d_in[23];
  const float* bc    = (const float*)d_in[24];
  const float* Wa1   = (const float*)d_in[25];
  const float* ba1   = (const float*)d_in[26];
  const float* Wa2   = (const float*)d_in[27];
  const float* ba2   = (const float*)d_in[28];
  const int* entities = (const int*)d_in[32];
  const int* locals_  = (const int*)d_in[33];
  const int* worlds   = (const int*)d_in[34];
  const int* commands = (const int*)d_in[35];
  float* out = (float*)d_out;

  char* p = (char*)d_ws;
  auto alloc = [&](size_t bytes)->char*{ char* r = p; p += (bytes + 255) & ~(size_t)255; return r; };

  h16*  emb_pad = (h16*) alloc(20000L*320*2);
  h16*  Wprjt   = (h16*) alloc(300L*512*2);
  h16*  dnWifH  = (h16*) alloc(768L*512*2);
  h16*  dnWibH  = (h16*) alloc(768L*512*2);
  h16*  Wcombf  = (h16*) alloc(768L*320*2);
  h16*  Wcombb  = (h16*) alloc(768L*320*2);
  h16*  dnWhfH  = (h16*) alloc(768L*256*2);
  h16*  dnWhbH  = (h16*) alloc(768L*256*2);
  int*  ids_all = (int*) alloc(180224L*4);
  float* h_f    = (float*)alloc(11264L*256*4);
  float* h_b    = (float*)alloc(11264L*256*4);
  h16*  hh_f    = (h16*) alloc(11264L*256*2);
  h16*  hh_b    = (h16*) alloc(11264L*256*2);
  float* Gh     = (float*)alloc(11264L*768*4);
  float* Gx     = (float*)alloc(11264L*768*4);
  h16*  Hout    = (h16*) alloc(11264L*512*2);
  h16*  wsum    = (h16*) alloc(1024L*512*2);
  h16*  Wet     = (h16*) alloc(512L*512*2);
  h16*  Wlt     = (h16*) alloc(512L*512*2);
  h16*  Wwt     = (h16*) alloc(512L*512*2);
  h16*  WgH     = (h16*) alloc(512L*512*2);
  float* t1     = (float*)alloc(1024L*512*4);
  float* t2     = (float*)alloc(1024L*512*4);
  float* t3     = (float*)alloc(1024L*512*4);
  h16*  egp     = (h16*) alloc(1024L*512*2);
  h16*  GEh     = (h16*) alloc(1024L*512*2);
  h16*  dWifH   = (h16*) alloc(1536L*512*2);
  h16*  dWibH   = (h16*) alloc(1536L*512*2);
  h16*  dWhfH   = (h16*) alloc(1536L*512*2);
  h16*  dWhbH   = (h16*) alloc(1536L*512*2);
  float* xgf    = (float*)alloc(1024L*1536*4);
  float* xgb    = (float*)alloc(1024L*1536*4);
  float* dh_f   = (float*)alloc(16L*512*4);
  float* dh_b   = (float*)alloc(16L*512*4);
  h16*  dhh_f   = (h16*) alloc(16L*512*2);
  h16*  dhh_b   = (h16*) alloc(16L*512*2);
  float* Ghd_f  = (float*)alloc(16L*1536*4);
  float* Ghd_b  = (float*)alloc(16L*1536*4);
  h16*  Wa1a    = (h16*) alloc(1024L*512*2);
  h16*  Wa1b    = (h16*) alloc(1024L*512*2);
  float* cpart  = (float*)alloc(1024L*1024*4);
  float* gpart  = (float*)alloc(1024L*1024*4);
  float* partial= (float*)alloc(65536L*4);

  auto gemm = [&](const h16* A,int lda,const int* gidx,int gstride,
                  const h16* B,int ldb,float* C,int ldc,h16* C2,int ldc2,
                  int M,int N,int K,const float* bias,int act){
    dim3 g(DIV_UP(N,BN), DIV_UP(M,BM));
    k_gemm_f16<<<g, dim3(256), 0, stream>>>(A,lda,gidx,gstride,B,ldb,C,ldc,C2,ldc2,M,N,K,bias,act);
  };
  auto conv = [&](const float* in,long in_ld,long in_off,h16* o,long out_ld,long rows,long cols){
    long n = rows*cols;
    k_conv_stride<<<DIV_UP(n,256),256,0,stream>>>(in,in_ld,in_off,o,out_ld,rows,cols);
  };
  auto transp = [&](const float* in,h16* o,int rows,int cols,float addc){
    k_transpose<<<DIV_UP((long)rows*cols,256),256,0,stream>>>(in,o,rows,cols,addc);
  };
  auto zf = [&](float* q,long n){ k_zero_f32<<<DIV_UP(n,256),256,0,stream>>>(q,n); };
  auto zh = [&](h16* q,long n){ k_zero_h16<<<DIV_UP(n,256),256,0,stream>>>(q,n); };

  // ---- weight prep ----
  k_emb_pad<<<DIV_UP(20000L*320,256),256,0,stream>>>(emb, emb_pad);
  transp(Wprj, Wprjt, 512, 300, 0.f);                 // [e][h]
  conv(dnWif, 512,0, dnWifH, 512, 768, 512);
  conv(dnWib, 512,0, dnWibH, 512, 768, 512);
  conv(dnWhf, 256,0, dnWhfH, 256, 768, 256);
  conv(dnWhb, 256,0, dnWhbH, 256, 768, 256);
  conv(Wg,    512,0, WgH,    512, 512, 512);
  conv(dWif,  512,0, dWifH,  512, 1536, 512);
  conv(dWib,  512,0, dWibH,  512, 1536, 512);
  conv(dWhf,  512,0, dWhfH,  512, 1536, 512);
  conv(dWhb,  512,0, dWhbH,  512, 1536, 512);
  conv(Wa1, 1024,  0, Wa1a, 512, 1024, 512);
  conv(Wa1, 1024,512, Wa1b, 512, 1024, 512);
  transp(W_e, Wet, 512, 512, 1.f);                    // (1 + W_e), right-multiply form
  transp(W_l, Wlt, 512, 512, 0.f);
  transp(W_w, Wwt, 512, 512, 0.f);
  // Wcomb = Wif @ Wprj  (768 x 300, padded ld 320)
  zh(Wcombf, 768L*320); zh(Wcombb, 768L*320);
  gemm(dnWifH,512,nullptr,0, Wprjt,512, nullptr,0, Wcombf,320, 768,300,512, nullptr,0);
  gemm(dnWibH,512,nullptr,0, Wprjt,512, nullptr,0, Wcombb,320, 768,300,512, nullptr,0);

  // ---- dn BiGRU over all 11264 sequences (len 16) ----
  k_pack_ids<<<DIV_UP(180224L,256),256,0,stream>>>(entities, locals_, worlds, commands, ids_all);
  zf(h_f, 11264L*256); zf(h_b, 11264L*256);
  zh(hh_f, 11264L*256); zh(hh_b, 11264L*256);
  for (int t = 0; t < 16; ++t) {
    // forward, token t
    gemm(emb_pad,320, ids_all + t,16, Wcombf,320, Gx,768, nullptr,0, 11264,768,320, dnbif,0);
    gemm(hh_f,256, nullptr,0, dnWhfH,256, Gh,768, nullptr,0, 11264,768,256, dnbhf,0);
    k_gru_update<<<DIV_UP(11264L*256,256),256,0,stream>>>(Gh,768, Gx,768, 1,0, h_f,hh_f, 11264,256);
    // backward, token 15-t
    int tok = 15 - t;
    gemm(emb_pad,320, ids_all + tok,16, Wcombb,320, Gx,768, nullptr,0, 11264,768,320, dnbib,0);
    gemm(hh_b,256, nullptr,0, dnWhbH,256, Gh,768, nullptr,0, 11264,768,256, dnbhb,0);
    k_gru_update<<<DIV_UP(11264L*256,256),256,0,stream>>>(Gh,768, Gx,768, 1,0, h_b,hh_b, 11264,256);
  }
  k_concat_h<<<DIV_UP(11264L*512,256),256,0,stream>>>(h_f, h_b, Hout);

  // ---- graph mixing ----
  k_wsum<<<DIV_UP(1024L*512,256),256,0,stream>>>(Hout, wsum);
  gemm(Hout,             512, nullptr,0, Wet,512, t1,512, nullptr,0, 1024,512,512, nullptr,0);
  gemm(Hout + 1024L*512, 512, nullptr,0, Wlt,512, t2,512, nullptr,0, 1024,512,512, nullptr,0);
  gemm(wsum,             512, nullptr,0, Wwt,512, t3,512, nullptr,0, 1024,512,512, nullptr,0);
  k_combine<<<DIV_UP(1024L*512,256),256,0,stream>>>(t1, t2, t3, egp);
  gemm(egp,512, nullptr,0, WgH,512, nullptr,0, GEh,512, 1024,512,512, bg,1);   // graph_encoding

  // ---- diff BiGRU (16 seqs, len 64, hidden 512) ----
  gemm(GEh,512, nullptr,0, dWifH,512, xgf,1536, nullptr,0, 1024,1536,512, dbif,0);
  gemm(GEh,512, nullptr,0, dWibH,512, xgb,1536, nullptr,0, 1024,1536,512, dbib,0);
  zf(dh_f, 16L*512); zf(dh_b, 16L*512); zh(dhh_f, 16L*512); zh(dhh_b, 16L*512);
  for (int t = 0; t < 64; ++t) {
    gemm(dhh_f,512, nullptr,0, dWhfH,512, Ghd_f,1536, nullptr,0, 16,1536,512, dbhf,0);
    k_gru_update<<<32,256,0,stream>>>(Ghd_f,1536, xgf,1536, 64,t, dh_f,dhh_f, 16,512);
    int tok = 63 - t;
    gemm(dhh_b,512, nullptr,0, dWhbH,512, Ghd_b,1536, nullptr,0, 16,1536,512, dbhb,0);
    k_gru_update<<<32,256,0,stream>>>(Ghd_b,1536, xgb,1536, 64,tok, dh_b,dhh_b, 16,512);
  }
  k_value<<<16,256,0,stream>>>(dh_f, dh_b, Wc, bc, out + 1040);

  // ---- attention scorer (rank-structured rep @ Wa1^T) ----
  gemm(Hout + 10240L*512,512, nullptr,0, Wa1a,512, cpart,1024, nullptr,0, 1024,1024,512, nullptr,0);
  gemm(GEh,512,              nullptr,0, Wa1b,512, gpart,1024, nullptr,0, 1024,1024,512, nullptr,0);
  k_score<<<1024,256,0,stream>>>(cpart, gpart, ba1, Wa2, partial);
  k_score_reduce<<<4,256,0,stream>>>(partial, ba2, out);
  k_index<<<16,64,0,stream>>>(out, commands, out + 1024);
}
